// EGNNLayer_77884936946197
// MI455X (gfx1250) — matrix operand order repaired and checked by reading, not compiled, not run on hardware
//
#include <hip/hip_runtime.h>
#include <math.h>

typedef __attribute__((ext_vector_type(16))) _Float16 v16h;
typedef __attribute__((ext_vector_type(8)))  _Float16 v8h;
typedef __attribute__((ext_vector_type(8)))  float    v8f;

#define BB    4
#define NNODE 16384
#define EE    65536
#define DH    128
#define DM    64
#define CLIPV 10000.0f

// nan_to_num: nan->0, +/-inf -> +/-CLIP, normal values passed through
// (finite |v| > CLIP also clamp; impossible for this problem's value ranges).
// v_med3_f32 + v_cmp_o + v_cndmask = 3 VALU, branchless.
__device__ __forceinline__ float n2n(float v) {
    const float c = __builtin_amdgcn_fmed3f(v, -CLIPV, CLIPV);
    return (v == v) ? c : 0.0f;
}
__device__ __forceinline__ float silu(float v) {
    return v / (1.0f + __expf(-v));
}

// ---------------------------------------------------------------------------
// prep kernels: zero scatter buffers, convert weights to f16 transposed (N-major)
// ---------------------------------------------------------------------------
__global__ void egnn_zero_kernel(float* __restrict__ p, int n) {
    int i = blockIdx.x * blockDim.x + threadIdx.x;
    if (i < n) p[i] = 0.0f;
}

__global__ void egnn_prep_kernel(const float* __restrict__ pe_w1,
                                 const float* __restrict__ pe_w2,
                                 const float* __restrict__ ph_w1,
                                 const float* __restrict__ ph_w2,
                                 _Float16* __restrict__ w1t,   // [64][256]
                                 _Float16* __restrict__ w2t,   // [64][64]
                                 _Float16* __restrict__ w1ht,  // [128][192]
                                 _Float16* __restrict__ w2ht)  // [128][128]
{
    int i = blockIdx.x * blockDim.x + threadIdx.x;
    if (i < 16384) {                       // pe_w1 rows 0..255 (row 256 stays f32)
        int n = i >> 8, k = i & 255;
        w1t[i] = (_Float16)pe_w1[k * DM + n];
        return;
    }
    i -= 16384;
    if (i < 4096) {
        int n = i >> 6, k = i & 63;
        w2t[i] = (_Float16)pe_w2[k * DM + n];
        return;
    }
    i -= 4096;
    if (i < 24576) {
        int n = i / 192, k = i % 192;
        w1ht[i] = (_Float16)ph_w1[k * DH + n];
        return;
    }
    i -= 24576;
    if (i < 16384) {
        int n = i >> 7, k = i & 127;
        w2ht[i] = (_Float16)ph_w2[k * DH + n];
    }
}

// pack two contiguous float4 loads into 8 f16 lanes of an A fragment
__device__ __forceinline__ void pack8_n2n(v16h& a, int base, const float* p) {
    const float4 f0 = *(const float4*)(p);
    const float4 f1 = *(const float4*)(p + 4);
    a[base + 0] = (_Float16)n2n(f0.x);
    a[base + 1] = (_Float16)n2n(f0.y);
    a[base + 2] = (_Float16)n2n(f0.z);
    a[base + 3] = (_Float16)n2n(f0.w);
    a[base + 4] = (_Float16)n2n(f1.x);
    a[base + 5] = (_Float16)n2n(f1.y);
    a[base + 6] = (_Float16)n2n(f1.z);
    a[base + 7] = (_Float16)n2n(f1.w);
}
__device__ __forceinline__ void pack8_raw(v16h& a, int base, const float* p) {
    const float4 f0 = *(const float4*)(p);
    const float4 f1 = *(const float4*)(p + 4);
    a[base + 0] = (_Float16)f0.x;
    a[base + 1] = (_Float16)f0.y;
    a[base + 2] = (_Float16)f0.z;
    a[base + 3] = (_Float16)f0.w;
    a[base + 4] = (_Float16)f1.x;
    a[base + 5] = (_Float16)f1.y;
    a[base + 6] = (_Float16)f1.z;
    a[base + 7] = (_Float16)f1.w;
}

// ---------------------------------------------------------------------------
// Edge kernel: one wave32 handles a tile of 16 edges.
//   GEMM1: [16 x 256] x [256 x 64]  (h_i | h_j parts of pe_w1), dist2 row rank-1
//   GEMM2: [16 x 64]  x [64 x 64]   (pe_w2)
//   scatter: atomic add m_ij -> dh_msg[src], m_x*dir -> dx[src]
// ---------------------------------------------------------------------------
__global__ __launch_bounds__(128) void egnn_edge_kernel(
    const float* __restrict__ x, const float* __restrict__ h,
    const int* __restrict__ ei,
    const float* __restrict__ pe_w1, const float* __restrict__ pe_b1,
    const float* __restrict__ pe_b2, const float* __restrict__ px_w,
    const _Float16* __restrict__ w1t, const _Float16* __restrict__ w2t,
    float* __restrict__ dh_msg, float* __restrict__ dx)
{
    __shared__ __align__(16) float    s_dist2[4][16];
    __shared__ __align__(16) float    s_diff[4][48];
    __shared__ __align__(16) int      s_src[4][16];
    __shared__ __align__(16) _Float16 s_m[4][16 * DM];

    const int tid  = threadIdx.x;
    const int wave = tid >> 5;
    const int lane = tid & 31;
    const int m16  = lane & 15;
    const int half = lane >> 4;

    const int tile  = blockIdx.x * 4 + wave;
    const int ebase = tile * 16;
    const int b     = ebase / EE;          // E % 16 == 0, so tile is batch-uniform
    const int e     = (ebase % EE) + m16;  // per-lane edge (row m16)

    // --- per-edge geometry (both halves compute row m16 redundantly) ---
    const int src = ei[((size_t)b * EE + e) * 2 + 0];
    const int dst = ei[((size_t)b * EE + e) * 2 + 1];
    const float* xs = x + ((size_t)b * NNODE + src) * 3;
    const float* xd = x + ((size_t)b * NNODE + dst) * 3;
    const float d0 = xs[0] - xd[0];
    const float d1 = xs[1] - xd[1];
    const float d2 = xs[2] - xd[2];
    const float dist2 = fmaxf(d0 * d0 + d1 * d1 + d2 * d2, 1e-12f);
    if (half == 0) {
        s_dist2[wave][m16] = dist2;
        s_diff[wave][m16 * 3 + 0] = d0;
        s_diff[wave][m16 * 3 + 1] = d1;
        s_diff[wave][m16 * 3 + 2] = d2;
        s_src[wave][m16] = src;
    }
    __syncthreads();

    // --- GEMM1: m_pre = [h_i | h_j] @ pe_w1[0:256]  (f16 WMMA, f32 acc) ---
    v8f acc[4] = {};
#pragma unroll
    for (int part = 0; part < 2; ++part) {
        const int node = (part == 0) ? src : dst;
        const float* hrow = h + ((size_t)b * NNODE + node) * DH;
#pragma unroll
        for (int ks = 0; ks < 4; ++ks) {
            v16h a;
            pack8_n2n(a, 0, hrow + ks * 32 + half * 16);
            pack8_n2n(a, 8, hrow + ks * 32 + half * 16 + 8);
#pragma unroll
            for (int nt = 0; nt < 4; ++nt) {
                const int n = nt * 16 + m16;
                const v16h bf = *(const v16h*)(w1t + n * 256 + part * 128
                                               + ks * 32 + half * 16);
                acc[nt] = __builtin_amdgcn_wmma_f32_16x16x32_f16(
                    false, a, false, bf, (short)0, acc[nt], false, false);
            }
        }
    }

    // --- epilogue: + dist2 * pe_w1[256,:] + b1, SiLU, stage to LDS as f16 ---
    const float* w1last = pe_w1 + 256 * DM;
#pragma unroll
    for (int nt = 0; nt < 4; ++nt) {
        const int n = nt * 16 + m16;
        const float wl = w1last[n];
        const float b1 = pe_b1[n];
#pragma unroll
        for (int r = 0; r < 8; ++r) {
            const int row = r + half * 8;
            float mv = acc[nt][r] + s_dist2[wave][row] * wl + b1;
            s_m[wave][row * DM + n] = (_Float16)silu(mv);
        }
    }
    __syncthreads();

    // --- GEMM2: m_ij = silu(m @ pe_w2 + b2) ---
    v8f acc2[4] = {};
#pragma unroll
    for (int ks = 0; ks < 2; ++ks) {
        v16h a;
#pragma unroll
        for (int g = 0; g < 2; ++g) {
            const v8h t8 = *(const v8h*)(&s_m[wave][m16 * DM + ks * 32
                                                   + half * 16 + g * 8]);
#pragma unroll
            for (int i = 0; i < 8; ++i) a[8 * g + i] = t8[i];
        }
#pragma unroll
        for (int nt = 0; nt < 4; ++nt) {
            const int n = nt * 16 + m16;
            const v16h bf = *(const v16h*)(w2t + n * DM + ks * 32 + half * 16);
            acc2[nt] = __builtin_amdgcn_wmma_f32_16x16x32_f16(
                false, a, false, bf, (short)0, acc2[nt], false, false);
        }
    }

    // --- epilogue: n2n(silu(.)), scatter dh_msg, accumulate m_x partials ---
    float pr[8] = {};
#pragma unroll
    for (int nt = 0; nt < 4; ++nt) {
        const int n = nt * 16 + m16;
        const float b2 = pe_b2[n];
        const float pw = px_w[n];
#pragma unroll
        for (int r = 0; r < 8; ++r) {
            const int row = r + half * 8;
            const float mij = n2n(silu(acc2[nt][r] + b2));
            atomicAdd(&dh_msg[((size_t)b * NNODE + s_src[wave][row]) * DM + n], mij);
            pr[r] += mij * pw;
        }
    }

    // --- m_x dot: reduce across the 16 lanes of each half, then scatter dx ---
#pragma unroll
    for (int r = 0; r < 8; ++r) {
        float t = pr[r];
        t += __shfl_xor(t, 1, 32);
        t += __shfl_xor(t, 2, 32);
        t += __shfl_xor(t, 4, 32);
        t += __shfl_xor(t, 8, 32);
        if (m16 == r) {
            const int   row = half * 8 + r;
            const float dd2 = s_dist2[wave][row];
            const float inv = 1.0f / (sqrtf(dd2) + 1e-8f);
            const int   sn  = s_src[wave][row];
            float* dxp = &dx[((size_t)b * NNODE + sn) * 3];
            atomicAdd(&dxp[0], n2n(t * s_diff[wave][row * 3 + 0] * inv));
            atomicAdd(&dxp[1], n2n(t * s_diff[wave][row * 3 + 1] * inv));
            atomicAdd(&dxp[2], n2n(t * s_diff[wave][row * 3 + 2] * inv));
        }
    }
}

// ---------------------------------------------------------------------------
// Node kernel: one wave32 handles a tile of 16 nodes.
//   GEMM1: [16 x 192] x [192 x 128]   (ph_w1 over [h | dh_msg])
//   GEMM2: [16 x 128] x [128 x 128]   (ph_w2), residual, n2n, write outputs
// ---------------------------------------------------------------------------
__global__ __launch_bounds__(128) void egnn_node_kernel(
    const float* __restrict__ x, const float* __restrict__ h,
    const float* __restrict__ ph_b1, const float* __restrict__ ph_b2,
    const _Float16* __restrict__ w1ht, const _Float16* __restrict__ w2ht,
    const float* __restrict__ dh_msg, const float* __restrict__ dx,
    float* __restrict__ out_x, float* __restrict__ out_h)
{
    __shared__ __align__(16) _Float16 s_t[4][16 * DH];

    const int tid  = threadIdx.x;
    const int wave = tid >> 5;
    const int lane = tid & 31;
    const int m16  = lane & 15;
    const int half = lane >> 4;

    const int tile  = blockIdx.x * 4 + wave;
    const int gbase = tile * 16;          // flat row index over B*N
    const int grow  = gbase + m16;        // per-lane row

    // --- GEMM1: K = 192 = 128 (h) + 64 (dh_msg) ---
    v8f acc[8] = {};
    const float* hrow = h      + (size_t)grow * DH;
    const float* drow = dh_msg + (size_t)grow * DM;
#pragma unroll
    for (int ks = 0; ks < 6; ++ks) {
        v16h a;
#pragma unroll
        for (int g = 0; g < 2; ++g) {
            const int k = ks * 32 + half * 16 + g * 8;
            const float* p = (k < DH) ? (hrow + k) : (drow + (k - DH));
            pack8_raw(a, 8 * g, p);     // reference does not n2n h_input
        }
#pragma unroll
        for (int nt = 0; nt < 8; ++nt) {
            const int n = nt * 16 + m16;
            const v16h bf = *(const v16h*)(w1ht + n * 192 + ks * 32 + half * 16);
            acc[nt] = __builtin_amdgcn_wmma_f32_16x16x32_f16(
                false, a, false, bf, (short)0, acc[nt], false, false);
        }
    }

    // --- epilogue: SiLU, stage to LDS f16 ---
#pragma unroll
    for (int nt = 0; nt < 8; ++nt) {
        const int n = nt * 16 + m16;
        const float b1 = ph_b1[n];
#pragma unroll
        for (int r = 0; r < 8; ++r) {
            const int row = r + half * 8;
            s_t[wave][row * DH + n] = (_Float16)silu(acc[nt][r] + b1);
        }
    }
    __syncthreads();

    // --- GEMM2: K = 128 ---
    v8f acc2[8] = {};
#pragma unroll
    for (int ks = 0; ks < 4; ++ks) {
        v16h a;
#pragma unroll
        for (int g = 0; g < 2; ++g) {
            const v8h t8 = *(const v8h*)(&s_t[wave][m16 * DH + ks * 32
                                                    + half * 16 + g * 8]);
#pragma unroll
            for (int i = 0; i < 8; ++i) a[8 * g + i] = t8[i];
        }
#pragma unroll
        for (int nt = 0; nt < 8; ++nt) {
            const int n = nt * 16 + m16;
            const v16h bf = *(const v16h*)(w2ht + n * DH + ks * 32 + half * 16);
            acc2[nt] = __builtin_amdgcn_wmma_f32_16x16x32_f16(
                false, a, false, bf, (short)0, acc2[nt], false, false);
        }
    }

    // --- epilogue: h_new = n2n(h + mlp + b2) ---
#pragma unroll
    for (int nt = 0; nt < 8; ++nt) {
        const int n = nt * 16 + m16;
        const float b2 = ph_b2[n];
#pragma unroll
        for (int r = 0; r < 8; ++r) {
            const int row = r + half * 8;
            const size_t g = (size_t)(gbase + row);
            out_h[g * DH + n] = n2n(h[g * DH + n] + acc2[nt][r] + b2);
        }
    }

    // --- x_new = n2n(x + dx) ---
    if (half == 0) {
        const size_t g = (size_t)grow;
#pragma unroll
        for (int c = 0; c < 3; ++c)
            out_x[g * 3 + c] = n2n(x[g * 3 + c] + dx[g * 3 + c]);
    }
}

// ---------------------------------------------------------------------------
extern "C" void kernel_launch(void* const* d_in, const int* in_sizes, int n_in,
                              void* d_out, int out_size, void* d_ws, size_t ws_size,
                              hipStream_t stream) {
    const float* x     = (const float*)d_in[0];
    const float* h     = (const float*)d_in[1];
    const int*   ei    = (const int*)d_in[2];
    const float* pe_w1 = (const float*)d_in[3];
    const float* pe_b1 = (const float*)d_in[4];
    const float* pe_w2 = (const float*)d_in[5];
    const float* pe_b2 = (const float*)d_in[6];
    const float* px_w  = (const float*)d_in[7];
    const float* ph_w1 = (const float*)d_in[8];
    const float* ph_b1 = (const float*)d_in[9];
    const float* ph_w2 = (const float*)d_in[10];
    const float* ph_b2 = (const float*)d_in[11];

    // workspace carve-up (all regions naturally aligned, contiguous)
    char* ws = (char*)d_ws;
    float*    dh_msg = (float*)ws;                                  // B*N*64 f32 = 16 MB
    float*    dxbuf  = (float*)(ws + (size_t)BB * NNODE * DM * 4);  // B*N*3 f32
    _Float16* w1t    = (_Float16*)(ws + (size_t)BB * NNODE * DM * 4
                                      + (size_t)BB * NNODE * 3 * 4);
    _Float16* w2t    = w1t  + 64 * 256;
    _Float16* w1ht   = w2t  + 64 * 64;
    _Float16* w2ht   = w1ht + 128 * 192;

    float* out_x = (float*)d_out;
    float* out_h = out_x + (size_t)BB * NNODE * 3;

    // 1) zero scatter accumulators (dh_msg and dx are contiguous)
    const int zn = BB * NNODE * DM + BB * NNODE * 3;
    egnn_zero_kernel<<<(zn + 255) / 256, 256, 0, stream>>>(dh_msg, zn);

    // 2) convert weights to f16 transposed
    egnn_prep_kernel<<<(61440 + 255) / 256, 256, 0, stream>>>(
        pe_w1, pe_w2, ph_w1, ph_w2, w1t, w2t, w1ht, w2ht);

    // 3) edge MLP + scatter: B*E/16 = 16384 wave-tiles, 4 waves/block
    egnn_edge_kernel<<<4096, 128, 0, stream>>>(
        x, h, ei, pe_w1, pe_b1, pe_b2, px_w, w1t, w2t, dh_msg, dxbuf);

    // 4) node MLP + residual: B*N/16 = 4096 wave-tiles, 4 waves/block
    egnn_node_kernel<<<1024, 128, 0, stream>>>(
        x, h, ph_b1, ph_b2, w1ht, w2ht, dh_msg, dxbuf, out_x, out_h);
}
